// BGCEncoder_84310208020493
// MI455X (gfx1250) — compile-verified, matrix operand order checked by reading, not run on hardware
//
#include <hip/hip_runtime.h>
#include <math.h>

#define B_  16
#define S_  1024
#define E_  1280
#define G_  3072
#define D_  512
#define H_  8
#define HD_ 64

typedef __attribute__((ext_vector_type(16))) __bf16 v16bf;
typedef __attribute__((ext_vector_type(8)))  float  v8f;

__device__ __forceinline__ v8f wmma_bf16(v16bf a, v16bf b, v8f c) {
    // D = A(16x32 bf16) * B(32x16 bf16) + C(16x16 f32)
    return __builtin_amdgcn_wmma_f32_16x16x32_bf16(
        false, a, false, b, (short)0, c, false, false);
}

__device__ __forceinline__ float gelu_exact(float v) {
    return 0.5f * v * (1.0f + erff(v * 0.70710678118654752f));
}

// gfx1250 async copy global -> LDS (ASYNCcnt-tracked, no VGPR transit).
// GVS mode: 64-bit SGPR base + 32-bit per-lane byte offset; vdst = LDS byte addr.
__device__ __forceinline__ void async_b128(unsigned lds, unsigned goff, const void* base) {
    asm volatile("global_load_async_to_lds_b128 %0, %1, %2"
                 :: "v"(lds), "v"(goff), "s"(base) : "memory");
}
__device__ __forceinline__ void wait_async0() {
    asm volatile("s_wait_asynccnt 0" ::: "memory");
}

// -------------------------------------------------------------------------
// fp32 -> bf16 conversion pass (memory bound; done once per tensor)
// -------------------------------------------------------------------------
__global__ __launch_bounds__(256)
void f32_to_bf16_kernel(const float* __restrict__ in, __bf16* __restrict__ out, int n)
{
    const int i = (blockIdx.x * blockDim.x + threadIdx.x) * 4;
    if (i + 3 < n) {
        const float4 v = *(const float4*)(in + i);
        out[i + 0] = (__bf16)v.x;
        out[i + 1] = (__bf16)v.y;
        out[i + 2] = (__bf16)v.z;
        out[i + 3] = (__bf16)v.w;
    } else {
        for (int j = i; j < n; ++j) out[j] = (__bf16)in[j];
    }
}

// -------------------------------------------------------------------------
// GEMM: C[M,N] = act(A_bf16[M,K] @ W_bf16[N,K]^T + bias) [* keep(row)]
// Block tile 128x64, 256 threads = 8 waves (4M x 2N), wave tile 32x32.
// Tiles staged via global_load_async_to_lds_b128, double-buffered in LDS.
// WF: write fp32 C; WB: write bf16 mirror.
// -------------------------------------------------------------------------
template <int ACT, bool MASK, bool WF, bool WB>
__global__ __launch_bounds__(256)
void gemm_bf16_async(const __bf16* __restrict__ A, const __bf16* __restrict__ W,
                     const float* __restrict__ bias, float* __restrict__ Cf,
                     __bf16* __restrict__ Cb, int M, int N, int K,
                     const bool* __restrict__ mask)
{
    // row stride 40 bf16 = 80 bytes (16B aligned chunks at row*80 + q*16)
    __shared__ __attribute__((aligned(16))) __bf16 As[2][128][40];
    __shared__ __attribute__((aligned(16))) __bf16 Ws[2][64][40];

    const int tid  = threadIdx.x;
    const int lane = tid & 31;
    const int wave = tid >> 5;
    const int lr   = lane & 15;
    const int hi   = lane >> 4;
    const int wm   = wave >> 1;      // 0..3
    const int wn   = wave & 1;       // 0..1

    const int blkM = blockIdx.y * 128;
    const int blkN = blockIdx.x * 64;

    const v8f zero = {};
    v8f acc[2][2];
    acc[0][0] = zero; acc[0][1] = zero; acc[1][0] = zero; acc[1][1] = zero;

    // async copy mapping: each thread moves three 16-byte chunks per K-step
    // (two for the 128x32 A tile, one for the 64x32 W tile)
    const int rowA0 = tid >> 2;          // 0..63
    const int rowA1 = 64 + rowA0;        // 64..127
    const int qq    = tid & 3;           // 16-byte quarter within a 64B row
    const unsigned asBase = (unsigned)(size_t)&As[0][0][0];
    const unsigned wsBase = (unsigned)(size_t)&Ws[0][0][0];
    const unsigned ldsA0  = (unsigned)(rowA0 * 80 + qq * 16);
    const unsigned ldsA1  = (unsigned)(rowA1 * 80 + qq * 16);
    const unsigned ldsW   = (unsigned)(rowA0 * 80 + qq * 16);
    unsigned goffA0 = (unsigned)(((blkM + rowA0) * K + qq * 8) * 2);
    unsigned goffA1 = (unsigned)(((blkM + rowA1) * K + qq * 8) * 2);
    unsigned goffW  = (unsigned)(((blkN + rowA0) * K + qq * 8) * 2);

    const int nk = K >> 5;

    // prologue: stage tile 0 into buffer 0
    async_b128(asBase + ldsA0, goffA0, A);
    async_b128(asBase + ldsA1, goffA1, A);
    async_b128(wsBase + ldsW,  goffW,  W);
    goffA0 += 64; goffA1 += 64; goffW += 64;
    wait_async0();
    __syncthreads();

    for (int t = 0; t < nk; ++t) {
        const int cur = t & 1;
        const int nxt = cur ^ 1;
        if (t + 1 < nk) {       // overlap: stage tile t+1 while computing tile t
            async_b128(asBase + (unsigned)(nxt * 10240) + ldsA0, goffA0, A);
            async_b128(asBase + (unsigned)(nxt * 10240) + ldsA1, goffA1, A);
            async_b128(wsBase + (unsigned)(nxt * 5120)  + ldsW,  goffW,  W);
            goffA0 += 64; goffA1 += 64; goffW += 64;
        }

        // A fragments: lane holds row lane&15, K-half lane>>4 (ISA 16-bit A layout)
        v16bf af[2], bfr[2];
        #pragma unroll
        for (int mf = 0; mf < 2; ++mf) {
            const int row = wm * 32 + mf * 16 + lr;
            #pragma unroll
            for (int r = 0; r < 8; ++r) {
                const int kk = ((r & 4) << 2) + hi * 8 + (r & 3) * 2;
                af[mf][2 * r]     = As[cur][row][kk];
                af[mf][2 * r + 1] = As[cur][row][kk + 1];
            }
        }
        // B fragments (B[k][n] = W[n][k]): lane holds column n, mirrored K packing
        #pragma unroll
        for (int nf = 0; nf < 2; ++nf) {
            const int col = wn * 32 + nf * 16 + lr;
            #pragma unroll
            for (int r = 0; r < 8; ++r) {
                const int kk = ((r & 4) << 2) + hi * 8 + (r & 3) * 2;
                bfr[nf][2 * r]     = Ws[cur][col][kk];
                bfr[nf][2 * r + 1] = Ws[cur][col][kk + 1];
            }
        }
        #pragma unroll
        for (int mf = 0; mf < 2; ++mf)
            #pragma unroll
            for (int nf = 0; nf < 2; ++nf)
                acc[mf][nf] = wmma_bf16(af[mf], bfr[nf], acc[mf][nf]);

        wait_async0();       // tile t+1 landed
        __syncthreads();     // everyone done reading tile t; safe to overwrite
    }

    // epilogue: bias + activation + optional row mask
    // C/D layout: lane col = lane&15, rows = (lane>>4)*8 + r
    #pragma unroll
    for (int mf = 0; mf < 2; ++mf) {
        #pragma unroll
        for (int nf = 0; nf < 2; ++nf) {
            const int col = blkN + wn * 32 + nf * 16 + lr;
            const float bval = bias[col];
            #pragma unroll
            for (int r = 0; r < 8; ++r) {
                const int row = blkM + wm * 32 + mf * 16 + hi * 8 + r;
                float v = acc[mf][nf][r] + bval;
                if (ACT == 1) v = gelu_exact(v);
                if (MASK)     v = mask[row] ? 0.0f : v;
                if (WF) Cf[(size_t)row * N + col] = v;
                if (WB) Cb[(size_t)row * N + col] = (__bf16)v;
            }
        }
    }
}

// -------------------------------------------------------------------------
// RoPE, applied in place to [B,S,D] with per-head 64-dim rotation
// -------------------------------------------------------------------------
__global__ __launch_bounds__(256)
void rope_kernel(float* __restrict__ x)
{
    const int p = blockIdx.x * blockDim.x + threadIdx.x;
    if (p >= B_ * S_ * H_ * 32) return;
    const int i    = p & 31;
    const int rest = p >> 5;             // (b*S+s)*H + h
    const int h    = rest & (H_ - 1);
    const int bs   = rest >> 3;          // b*S + s
    const int s    = bs & (S_ - 1);
    const float freq = (float)s * __expf(-(float)i * 0.28782313662425575f); // ln(1e4)/32
    float sn, cs;
    __sincosf(freq, &sn, &cs);
    float* ptr = x + (size_t)bs * D_ + h * HD_ + 2 * i;
    const float x0 = ptr[0], x1 = ptr[1];
    ptr[0] = x0 * cs - x1 * sn;
    ptr[1] = x1 * cs + x0 * sn;
}

// -------------------------------------------------------------------------
// Flash attention with structural bias:
//   s = (q.k^T + beta_h * b.k^T) / 8, online softmax, ctx = P.V (bf16 out)
// Block = 128 threads (4 waves); block owns (b, h, 64 q-rows).
// -------------------------------------------------------------------------
__global__ __launch_bounds__(128)
void attn_flash(const float* __restrict__ q, const float* __restrict__ kg,
                const float* __restrict__ vg, const float* __restrict__ bp,
                const float* __restrict__ beta, const bool* __restrict__ mask,
                __bf16* __restrict__ ctx)
{
    __shared__ __bf16 Ks[32][HD_ + 4];
    __shared__ __bf16 Vs[32][HD_ + 4];
    __shared__ __bf16 Ps[4][16][40];     // per-wave P staging (C-layout -> A-layout)
    __shared__ float  mb[32];

    const int tid  = threadIdx.x;
    const int lane = tid & 31;
    const int w    = tid >> 5;
    const int lr   = lane & 15;
    const int hi   = lane >> 4;

    const int qt = blockIdx.x;           // 0..15
    const int h  = blockIdx.y;           // 0..7
    const int b  = blockIdx.z;           // 0..15
    const int qbase = qt * 64 + w * 16;
    const float betah = beta[h];

    // q and bias-projection A-fragments: 16x64 -> two 16x32 frags each
    v16bf qa[2], ba[2];
    {
        const float* qrow = q  + ((size_t)b * S_ + qbase + lr) * D_ + h * HD_;
        const float* brow = bp + ((size_t)b * S_ + qbase + lr) * D_ + h * HD_;
        #pragma unroll
        for (int kf = 0; kf < 2; ++kf)
            #pragma unroll
            for (int r = 0; r < 8; ++r) {
                const int kk = kf * 32 + ((r & 4) << 2) + hi * 8 + (r & 3) * 2;
                qa[kf][2 * r]     = (__bf16)qrow[kk];
                qa[kf][2 * r + 1] = (__bf16)qrow[kk + 1];
                ba[kf][2 * r]     = (__bf16)brow[kk];
                ba[kf][2 * r + 1] = (__bf16)brow[kk + 1];
            }
    }

    float mrow[8], lrow[8];
    const v8f zero = {};
    v8f o[4];
    #pragma unroll
    for (int r = 0; r < 8; ++r) { mrow[r] = -INFINITY; lrow[r] = 0.0f; }
    o[0] = zero; o[1] = zero; o[2] = zero; o[3] = zero;

    for (int kt = 0; kt < S_; kt += 32) {
        {
            const int key = tid >> 2;
            const int d0  = (tid & 3) * 16;
            const float* ksrc = kg + ((size_t)b * S_ + kt + key) * D_ + h * HD_ + d0;
            const float* vsrc = vg + ((size_t)b * S_ + kt + key) * D_ + h * HD_ + d0;
            #pragma unroll
            for (int j = 0; j < 16; ++j) {
                Ks[key][d0 + j] = (__bf16)ksrc[j];
                Vs[key][d0 + j] = (__bf16)vsrc[j];
            }
        }
        if (tid < 32) mb[tid] = mask[(size_t)b * S_ + kt + tid] ? -INFINITY : 0.0f;
        __syncthreads();

        v16bf kb[2][2];
        #pragma unroll
        for (int sub = 0; sub < 2; ++sub)
            #pragma unroll
            for (int kf = 0; kf < 2; ++kf)
                #pragma unroll
                for (int r = 0; r < 8; ++r) {
                    const int dd = kf * 32 + ((r & 4) << 2) + hi * 8 + (r & 3) * 2;
                    kb[sub][kf][2 * r]     = Ks[sub * 16 + lr][dd];
                    kb[sub][kf][2 * r + 1] = Ks[sub * 16 + lr][dd + 1];
                }

        float sc[2][8];
        #pragma unroll
        for (int sub = 0; sub < 2; ++sub) {
            v8f sa = {};
            v8f sb = {};
            sa = wmma_bf16(qa[0], kb[sub][0], sa);
            sa = wmma_bf16(qa[1], kb[sub][1], sa);
            sb = wmma_bf16(ba[0], kb[sub][0], sb);
            sb = wmma_bf16(ba[1], kb[sub][1], sb);
            #pragma unroll
            for (int r = 0; r < 8; ++r)
                sc[sub][r] = (sa[r] + betah * sb[r]) * 0.125f + mb[sub * 16 + lr];
        }

        float nm[8], alpha[8];
        #pragma unroll
        for (int r = 0; r < 8; ++r) {
            float v = fmaxf(sc[0][r], sc[1][r]);
            #pragma unroll
            for (int off = 1; off < 16; off <<= 1)
                v = fmaxf(v, __shfl_xor(v, off, 32));
            nm[r]    = fmaxf(mrow[r], v);
            alpha[r] = __expf(mrow[r] - nm[r]);
            mrow[r]  = nm[r];
        }
        #pragma unroll
        for (int r = 0; r < 8; ++r) {
            const float p0 = __expf(sc[0][r] - nm[r]);
            const float p1 = __expf(sc[1][r] - nm[r]);
            Ps[w][hi * 8 + r][lr]      = (__bf16)p0;
            Ps[w][hi * 8 + r][16 + lr] = (__bf16)p1;
            float rs = p0 + p1;
            #pragma unroll
            for (int off = 1; off < 16; off <<= 1)
                rs += __shfl_xor(rs, off, 32);
            lrow[r] = lrow[r] * alpha[r] + rs;
        }
        #pragma unroll
        for (int nf = 0; nf < 4; ++nf)
            #pragma unroll
            for (int r = 0; r < 8; ++r)
                o[nf][r] *= alpha[r];

        v16bf pa;
        #pragma unroll
        for (int r = 0; r < 8; ++r) {
            const int kk = ((r & 4) << 2) + hi * 8 + (r & 3) * 2;
            pa[2 * r]     = Ps[w][lr][kk];
            pa[2 * r + 1] = Ps[w][lr][kk + 1];
        }
        #pragma unroll
        for (int nf = 0; nf < 4; ++nf) {
            v16bf vb;
            #pragma unroll
            for (int r = 0; r < 8; ++r) {
                const int kk = ((r & 4) << 2) + hi * 8 + (r & 3) * 2;
                vb[2 * r]     = Vs[kk][nf * 16 + lr];
                vb[2 * r + 1] = Vs[kk + 1][nf * 16 + lr];
            }
            o[nf] = wmma_bf16(pa, vb, o[nf]);
        }
        __syncthreads();
    }

    // normalize and write bf16 ctx [B,S,D] (only consumer is the Wo GEMM)
    #pragma unroll
    for (int nf = 0; nf < 4; ++nf)
        #pragma unroll
        for (int r = 0; r < 8; ++r) {
            const int srow = qbase + hi * 8 + r;
            ctx[((size_t)b * S_ + srow) * D_ + h * HD_ + nf * 16 + lr] =
                (__bf16)(o[nf][r] / lrow[r]);
        }
}

// -------------------------------------------------------------------------
// out = LN(a + res) * gamma + beta  [ * keep ]; optional bf16 mirror
// -------------------------------------------------------------------------
__global__ __launch_bounds__(256)
void ln_residual(const float* __restrict__ a, const float* __restrict__ res,
                 const float* __restrict__ g, const float* __restrict__ bb,
                 float* __restrict__ out, __bf16* __restrict__ outb,
                 const bool* __restrict__ mask, int applyMask, int writeB)
{
    __shared__ float s1[256], s2[256];
    const int row = blockIdx.x;
    const int tid = threadIdx.x;
    const size_t base = (size_t)row * D_;
    const float v0 = a[base + tid]       + res[base + tid];
    const float v1 = a[base + tid + 256] + res[base + tid + 256];
    s1[tid] = v0 + v1;
    s2[tid] = v0 * v0 + v1 * v1;
    __syncthreads();
    for (int off = 128; off > 0; off >>= 1) {
        if (tid < off) { s1[tid] += s1[tid + off]; s2[tid] += s2[tid + off]; }
        __syncthreads();
    }
    const float mean = s1[0] * (1.0f / D_);
    const float var  = s2[0] * (1.0f / D_) - mean * mean;
    const float rstd = rsqrtf(var + 1e-5f);
    const float keep = (applyMask && mask[row]) ? 0.0f : 1.0f;
    const float r0 = ((v0 - mean) * rstd * g[tid]       + bb[tid])       * keep;
    const float r1 = ((v1 - mean) * rstd * g[tid + 256] + bb[tid + 256]) * keep;
    out[base + tid]       = r0;
    out[base + tid + 256] = r1;
    if (writeB) {
        outb[base + tid]       = (__bf16)r0;
        outb[base + tid + 256] = (__bf16)r1;
    }
}

// -------------------------------------------------------------------------
extern "C" void kernel_launch(void* const* d_in, const int* in_sizes, int n_in,
                              void* d_out, int out_size, void* d_ws, size_t ws_size,
                              hipStream_t stream)
{
    (void)in_sizes; (void)n_in; (void)out_size; (void)ws_size;

    const float* pros      = (const float*)d_in[0];
    const bool*  mask      = (const bool*) d_in[1];
    const float* structure = (const float*)d_in[2];
    const float* We = (const float*)d_in[3];  const float* be = (const float*)d_in[4];
    const float* Wg = (const float*)d_in[5];  const float* bg = (const float*)d_in[6];
    const float* Wq = (const float*)d_in[7];  const float* bq = (const float*)d_in[8];
    const float* Wk = (const float*)d_in[9];  const float* bk = (const float*)d_in[10];
    const float* Wv = (const float*)d_in[11]; const float* bv = (const float*)d_in[12];
    const float* Wb = (const float*)d_in[13]; const float* bbp= (const float*)d_in[14];
    const float* Wo = (const float*)d_in[15]; const float* bo = (const float*)d_in[16];
    const float* beta = (const float*)d_in[17];
    const float* W1 = (const float*)d_in[18]; const float* b1 = (const float*)d_in[19];
    const float* W2 = (const float*)d_in[20]; const float* b2 = (const float*)d_in[21];
    const float* g1 = (const float*)d_in[22]; const float* bn1 = (const float*)d_in[23];
    const float* g2 = (const float*)d_in[24]; const float* bn2 = (const float*)d_in[25];
    float* out = (float*)d_out;

    const int    M  = B_ * S_;
    const size_t MD = (size_t)M * D_;

    // byte-cursor workspace allocation (256B aligned)
    char* p = (char*)d_ws;
    auto alloc = [&](size_t bytes) -> void* {
        void* r = (void*)p; p += (bytes + 255) & ~(size_t)255; return r;
    };
    __bf16* WeB = (__bf16*)alloc((size_t)D_ * E_ * 2);
    __bf16* WgB = (__bf16*)alloc((size_t)D_ * G_ * 2);
    __bf16* WqB = (__bf16*)alloc((size_t)D_ * D_ * 2);
    __bf16* WkB = (__bf16*)alloc((size_t)D_ * D_ * 2);
    __bf16* WvB = (__bf16*)alloc((size_t)D_ * D_ * 2);
    __bf16* WbB = (__bf16*)alloc((size_t)D_ * D_ * 2);
    __bf16* WoB = (__bf16*)alloc((size_t)D_ * D_ * 2);
    __bf16* W1B = (__bf16*)alloc((size_t)2 * D_ * D_ * 2);
    __bf16* W2B = (__bf16*)alloc((size_t)D_ * 2 * D_ * 2);
    __bf16* prosB = (__bf16*)alloc((size_t)M * E_ * 2);   // later reused as ctxB
    __bf16* stInB = (__bf16*)alloc((size_t)M * G_ * 2);   // later reused as ao/h/hB
    float*  x     = (float*) alloc(MD * 4);
    __bf16* xB    = (__bf16*)alloc(MD * 2);
    __bf16* stB   = (__bf16*)alloc(MD * 2);
    float*  qb    = (float*) alloc(MD * 4);
    float*  kbuf  = (float*) alloc(MD * 4);
    float*  vbuf  = (float*) alloc(MD * 4);
    float*  bpb   = (float*) alloc(MD * 4);
    __bf16* ff1B  = (__bf16*)alloc(2 * MD * 2);
    float*  ff2   = (float*) alloc(MD * 4);
    // aliases into dead regions:
    __bf16* ctxB = prosB;                            // 16.8MB into 41.9MB region
    float*  ao   = (float*)stInB;                    // 33.5MB
    float*  hbuf = (float*)((char*)stInB + MD * 4);  // 33.5MB
    __bf16* hB   = (__bf16*)((char*)stInB + 2 * MD * 4); // 16.8MB (region is 100.7MB)

    dim3 blk(256);
    auto cgrid = [](size_t n) { return dim3((unsigned)((n / 4 + 255) / 256)); };

    // one-time fp32 -> bf16 conversions (weights + raw inputs)
    f32_to_bf16_kernel<<<cgrid((size_t)D_ * E_),    blk, 0, stream>>>(We, WeB, D_ * E_);
    f32_to_bf16_kernel<<<cgrid((size_t)D_ * G_),    blk, 0, stream>>>(Wg, WgB, D_ * G_);
    f32_to_bf16_kernel<<<cgrid((size_t)D_ * D_),    blk, 0, stream>>>(Wq, WqB, D_ * D_);
    f32_to_bf16_kernel<<<cgrid((size_t)D_ * D_),    blk, 0, stream>>>(Wk, WkB, D_ * D_);
    f32_to_bf16_kernel<<<cgrid((size_t)D_ * D_),    blk, 0, stream>>>(Wv, WvB, D_ * D_);
    f32_to_bf16_kernel<<<cgrid((size_t)D_ * D_),    blk, 0, stream>>>(Wb, WbB, D_ * D_);
    f32_to_bf16_kernel<<<cgrid((size_t)D_ * D_),    blk, 0, stream>>>(Wo, WoB, D_ * D_);
    f32_to_bf16_kernel<<<cgrid((size_t)2 * D_ * D_),blk, 0, stream>>>(W1, W1B, 2 * D_ * D_);
    f32_to_bf16_kernel<<<cgrid((size_t)2 * D_ * D_),blk, 0, stream>>>(W2, W2B, 2 * D_ * D_);
    f32_to_bf16_kernel<<<cgrid((size_t)M * E_),     blk, 0, stream>>>(pros,      prosB, M * E_);
    f32_to_bf16_kernel<<<cgrid((size_t)M * G_),     blk, 0, stream>>>(structure, stInB, M * G_);

    // encoder pipeline
    gemm_bf16_async<1, true,  true,  true ><<<dim3(D_ / 64, M / 128), blk, 0, stream>>>(prosB, WeB, be,  x,       xB,    M, D_,     E_,     mask);
    gemm_bf16_async<1, false, false, true ><<<dim3(D_ / 64, M / 128), blk, 0, stream>>>(stInB, WgB, bg,  nullptr, stB,   M, D_,     G_,     nullptr);
    gemm_bf16_async<0, false, true,  false><<<dim3(D_ / 64, M / 128), blk, 0, stream>>>(xB,  WqB, bq,  qb,   nullptr, M, D_, D_, nullptr);
    gemm_bf16_async<0, false, true,  false><<<dim3(D_ / 64, M / 128), blk, 0, stream>>>(xB,  WkB, bk,  kbuf, nullptr, M, D_, D_, nullptr);
    gemm_bf16_async<0, false, true,  false><<<dim3(D_ / 64, M / 128), blk, 0, stream>>>(xB,  WvB, bv,  vbuf, nullptr, M, D_, D_, nullptr);
    gemm_bf16_async<0, false, true,  false><<<dim3(D_ / 64, M / 128), blk, 0, stream>>>(stB, WbB, bbp, bpb,  nullptr, M, D_, D_, nullptr);

    const int ropeTotal = B_ * S_ * H_ * 32;
    rope_kernel<<<(ropeTotal + 255) / 256, blk, 0, stream>>>(qb);
    rope_kernel<<<(ropeTotal + 255) / 256, blk, 0, stream>>>(kbuf);

    attn_flash<<<dim3(S_ / 64, H_, B_), dim3(128), 0, stream>>>(qb, kbuf, vbuf, bpb, beta, mask, ctxB);

    gemm_bf16_async<0, false, true,  false><<<dim3(D_ / 64, M / 128), blk, 0, stream>>>(ctxB, WoB, bo, ao, nullptr, M, D_, D_, nullptr);
    ln_residual<<<M, blk, 0, stream>>>(ao, x, g1, bn1, hbuf, hB, mask, 0, 1);
    gemm_bf16_async<1, false, false, true ><<<dim3((2 * D_) / 64, M / 128), blk, 0, stream>>>(hB,   W1B, b1, nullptr, ff1B, M, 2 * D_, D_,     nullptr);
    gemm_bf16_async<0, false, true,  false><<<dim3(D_ / 64,       M / 128), blk, 0, stream>>>(ff1B, W2B, b2, ff2,     nullptr, M, D_,  2 * D_, nullptr);
    ln_residual<<<M, blk, 0, stream>>>(ff2, hbuf, g2, bn2, out, nullptr, mask, 1, 0);
}
